// LKAN_5394478924095
// MI455X (gfx1250) — matrix operand order repaired and checked by reading, not compiled, not run on hardware
//
#include <hip/hip_runtime.h>
#include <hip/hip_bf16.h>

// ---------------------------------------------------------------------------
// LKAN fused kernel for MI455X (gfx1250, wave32, WMMA + async global->LDS)
//
// Problem: B=8, S=512, F=1024, U=1024, COEFFS=11
//   1) per-batch LayerNorm over (S,F)
//   2) base:   silu(xn) @ base_weight            (K=1024)
//   3) spline: bsplines(xn) @ spline_weight^T    (K=11264)
// Fused as one bf16 WMMA GEMM: M=4096, N=1024, K=12288, fed by
// GLOBAL_LOAD_ASYNC_TO_LDS_B128 double buffering (ASYNCcnt-tracked).
// ---------------------------------------------------------------------------

typedef __attribute__((ext_vector_type(16))) __bf16 v16bf;
typedef __attribute__((ext_vector_type(8)))  __bf16 v8bf;
typedef __attribute__((ext_vector_type(8)))  float  v8f;

union V16U { v16bf v; v8bf h[2]; };

#define NB       8
#define NS       512
#define NF       1024
#define NU       1024
#define NCOEF    11
#define KBASE    1024
#define KSPL     (NF * NCOEF)          // 11264
#define KTOT     (KBASE + KSPL)        // 12288
#define NM       (NB * NS)             // 4096
#define PER_B    (NS * NF)             // 524288

// Async 16-byte global -> LDS copy (no VGPR staging; tracked by ASYNCcnt).
// LDS operand: low 32 bits of a generic LDS pointer are the LDS byte offset
// (ISA 10.2 aperture rules).
__device__ __forceinline__ void async_copy_b128(const void* g, void* l) {
    asm volatile("global_load_async_to_lds_b128 %0, %1, off"
                 :: "v"((unsigned)(size_t)l),
                    "v"((unsigned long long)(size_t)g)
                 : "memory");
}
__device__ __forceinline__ void wait_asynccnt0() {
    asm volatile("s_wait_asynccnt 0x0" ::: "memory");
}

// ---------------------------------------------------------------------------
// Kernel 1: per-batch mean / rsqrt(var+eps).  8 blocks, 1024 threads each.
// ---------------------------------------------------------------------------
__global__ __launch_bounds__(1024) void kan_stats(const float* __restrict__ x,
                                                  float* __restrict__ stats) {
    const int b = blockIdx.x;
    const float* xb = x + (size_t)b * PER_B;
    float s = 0.f, sq = 0.f;
    for (int i = threadIdx.x; i < PER_B; i += 1024) {
        float v = xb[i];
        s += v; sq += v * v;
    }
    __shared__ float ss[1024];
    __shared__ float sq2[1024];
    ss[threadIdx.x] = s; sq2[threadIdx.x] = sq;
    __syncthreads();
    for (int off = 512; off > 0; off >>= 1) {
        if (threadIdx.x < off) {
            ss[threadIdx.x]  += ss[threadIdx.x + off];
            sq2[threadIdx.x] += sq2[threadIdx.x + off];
        }
        __syncthreads();
    }
    if (threadIdx.x == 0) {
        float mu  = ss[0] / (float)PER_B;
        float var = sq2[0] / (float)PER_B - mu * mu;   // biased variance
        stats[2 * b + 0] = mu;
        stats[2 * b + 1] = rsqrtf(var + 1e-5f);
    }
}

// ---------------------------------------------------------------------------
// Kernel 2: build fused A matrix [4096][12288] bf16:
//   A[r][f]                 = silu(LN(x))          (f < 1024)
//   A[r][1024 + f*11 + c]   = Cox-de-Boor basis c  (uniform knots)
// ---------------------------------------------------------------------------
__global__ __launch_bounds__(256) void kan_features(const float* __restrict__ x,
                                                    const float* __restrict__ gamma,
                                                    const float* __restrict__ beta,
                                                    const float* __restrict__ stats,
                                                    __bf16* __restrict__ Afeat) {
    const int idx = blockIdx.x * 256 + threadIdx.x;   // 0 .. 4194303
    const int f = idx & (NF - 1);
    const int r = idx >> 10;                          // row in [0, 4096)
    const int b = r >> 9;
    const int s = r & (NS - 1);

    const float mu = stats[2 * b + 0];
    const float rs = stats[2 * b + 1];
    const float xn = (x[idx] - mu) * rs * gamma[s * NF + f] + beta[s * NF + f];

    __bf16* arow = Afeat + (size_t)r * KTOT;
    // silu
    arow[f] = (__bf16)(xn / (1.f + __expf(-xn)));

    // Cox-de-Boor, uniform knots g(i) = -1.75 + 0.25*i, i=0..14
    float bas[14];
#pragma unroll
    for (int j = 0; j < 14; ++j) {
        float g0 = -1.75f + 0.25f * (float)j;
        bas[j] = (xn >= g0 && xn < g0 + 0.25f) ? 1.f : 0.f;
    }
#pragma unroll
    for (int k = 1; k <= 3; ++k) {
        const float inv = 4.f / (float)k;             // 1/(0.25*k)
#pragma unroll
        for (int j = 0; j < 14 - k; ++j) {
            float gj    = -1.75f + 0.25f * (float)j;
            float left  = (xn - gj) * inv;
            float right = ((gj + 0.25f * (float)(k + 1)) - xn) * inv;
            bas[j] = left * bas[j] + right * bas[j + 1];
        }
    }
    __bf16* srow = arow + KBASE + f * NCOEF;
#pragma unroll
    for (int c = 0; c < NCOEF; ++c) srow[c] = (__bf16)bas[c];
}

// ---------------------------------------------------------------------------
// Kernel 3: pack combined weight, transposed:  WcT[n][k], bf16, K contiguous.
//   k < 1024 : base_weight[k, n]      (contracted along its FIRST axis)
//   k >= 1024: spline_weight[n, k-1024]
// ---------------------------------------------------------------------------
__global__ __launch_bounds__(256) void kan_pack_w(const float* __restrict__ base_w,
                                                  const float* __restrict__ spline_w,
                                                  __bf16* __restrict__ WcT) {
    const size_t idx = (size_t)blockIdx.x * 256 + threadIdx.x;  // 0 .. 12582911
    const int k = (int)(idx % KTOT);
    const int n = (int)(idx / KTOT);
    float v = (k < KBASE) ? base_w[(size_t)k * NU + n]
                          : spline_w[(size_t)n * KSPL + (k - KBASE)];
    WcT[idx] = (__bf16)v;
}

// ---------------------------------------------------------------------------
// Kernel 4: bf16 WMMA GEMM.  C[4096][1024] = A[4096][12288] * WcT^T + bias.
// Block tile 128x128, KB=64, double-buffered LDS fed by async global->LDS.
// 8 waves -> each wave 32x64 subtile = 2x4 v_wmma_f32_16x16x32_bf16 / k-step.
// ---------------------------------------------------------------------------
__global__ __launch_bounds__(256) void kan_gemm(const __bf16* __restrict__ A,
                                                const __bf16* __restrict__ BT,
                                                const float* __restrict__ bias,
                                                float* __restrict__ C) {
    __shared__ __align__(16) __bf16 Atile[2][128 * 64];   // 2 x 16 KB
    __shared__ __align__(16) __bf16 Btile[2][128 * 64];   // 2 x 16 KB

    const int tid  = threadIdx.x;
    const int lane = tid & 31;
    const int w    = tid >> 5;          // wave 0..7
    const int wm   = w & 3;             // 4 waves along M
    const int wn   = w >> 2;            // 2 waves along N
    const int lrow = lane & 15;
    const int khalf = lane >> 4;        // 0 | 1

    const int bm = blockIdx.x;          // 0..31  (M tiles of 128)
    const int bn = blockIdx.y;          // 0..7   (N tiles of 128)

    v8f acc[2][4] = {};

    // Stage one 128x64 slab of A and BT into LDS buffer `buf` via async DMA.
    auto stage = [&](int buf, int kb) {
#pragma unroll
        for (int t = 0; t < 4; ++t) {
            const int chunk = tid + t * 256;       // 0..1023 16B chunks
            const int row   = chunk >> 3;          // 8 chunks per 64-elem row
            const int col   = (chunk & 7) * 8;
            async_copy_b128(&A[(size_t)(bm * 128 + row) * KTOT + kb + col],
                            &Atile[buf][row * 64 + col]);
            async_copy_b128(&BT[(size_t)(bn * 128 + row) * KTOT + kb + col],
                            &Btile[buf][row * 64 + col]);
        }
    };

    // Prologue: fill buffer 0.
    stage(0, 0);
    wait_asynccnt0();
    __syncthreads();

    int buf = 0;
    for (int kb = 0; kb < KTOT; kb += 64) {
        // Kick off next slab into the other buffer while we compute.
        if (kb + 64 < KTOT) stage(buf ^ 1, kb + 64);

        // --- two 16x16x32 k-steps on the current slab ---
#pragma unroll
        for (int ks = 0; ks < 64; ks += 32) {
            V16U afrag[2];
#pragma unroll
            for (int i = 0; i < 2; ++i) {
                const int r = wm * 32 + i * 16 + lrow;
                // A 16x32 bf16 layout: lanes0-15 K=[0..8)+[16..24), lanes16-31 +8
                afrag[i].h[0] = *reinterpret_cast<const v8bf*>(
                    &Atile[buf][r * 64 + ks + khalf * 8]);
                afrag[i].h[1] = *reinterpret_cast<const v8bf*>(
                    &Atile[buf][r * 64 + ks + 16 + khalf * 8]);
            }
            V16U bfrag[4];
#pragma unroll
            for (int j = 0; j < 4; ++j) {
                const int cidx = wn * 64 + j * 16 + lrow;
                // B 32x16 bf16 layout: lanes0-15 K=[0..8)+[16..24), lanes16-31 +8
                bfrag[j].h[0] = *reinterpret_cast<const v8bf*>(
                    &Btile[buf][cidx * 64 + ks + khalf * 8]);
                bfrag[j].h[1] = *reinterpret_cast<const v8bf*>(
                    &Btile[buf][cidx * 64 + ks + 16 + khalf * 8]);
            }
#pragma unroll
            for (int i = 0; i < 2; ++i)
#pragma unroll
                for (int j = 0; j < 4; ++j)
                    acc[i][j] = __builtin_amdgcn_wmma_f32_16x16x32_bf16(
                        /*neg_a=*/false, afrag[i].v,
                        /*neg_b=*/false, bfrag[j].v,
                        /*c_mod=*/(short)0, acc[i][j],
                        /*reuse_a=*/false, /*reuse_b=*/false);
        }

        // Next slab must be fully landed and everyone done reading this one.
        wait_asynccnt0();
        __syncthreads();
        buf ^= 1;
    }

    // --- epilogue: C layout VGPR e -> M = khalf*8 + e, N = lane&15; +bias ---
#pragma unroll
    for (int i = 0; i < 2; ++i) {
#pragma unroll
        for (int j = 0; j < 4; ++j) {
            const int ncol  = bn * 128 + wn * 64 + j * 16 + lrow;
            const int mbase = bm * 128 + wm * 32 + i * 16 + khalf * 8;
            const float bv  = bias[ncol];
#pragma unroll
            for (int e = 0; e < 8; ++e)
                C[(size_t)(mbase + e) * NU + ncol] = acc[i][j][e] + bv;
        }
    }
}

// ---------------------------------------------------------------------------
// Host launcher
// ---------------------------------------------------------------------------
extern "C" void kernel_launch(void* const* d_in, const int* in_sizes, int n_in,
                              void* d_out, int out_size, void* d_ws, size_t ws_size,
                              hipStream_t stream) {
    const float* x        = (const float*)d_in[0];
    const float* ln_gamma = (const float*)d_in[1];
    const float* ln_beta  = (const float*)d_in[2];
    const float* base_w   = (const float*)d_in[3];
    const float* base_b   = (const float*)d_in[4];
    const float* spline_w = (const float*)d_in[5];
    float* out = (float*)d_out;

    char* ws = (char*)d_ws;
    float*  stats = (float*)ws;                                       // 16 floats
    __bf16* Afeat = (__bf16*)(ws + 256);                              // 4096*12288 bf16
    __bf16* WcT   = (__bf16*)(ws + 256 + (size_t)NM * KTOT * 2);      // 1024*12288 bf16

    kan_stats<<<NB, 1024, 0, stream>>>(x, stats);

    kan_features<<<(NM * NF) / 256, 256, 0, stream>>>(x, ln_gamma, ln_beta, stats, Afeat);

    kan_pack_w<<<(int)(((size_t)NU * KTOT) / 256), 256, 0, stream>>>(base_w, spline_w, WcT);

    dim3 grid(NM / 128, NU / 128);   // 32 x 8
    kan_gemm<<<grid, 256, 0, stream>>>(Afeat, WcT, base_b, out);
}
